// SparseMedNeXtBlock_7086696039110
// MI455X (gfx1250) — matrix operand order repaired and checked by reading, not compiled, not run on hardware
//
#include <hip/hip_runtime.h>
#include <hip/hip_bf16.h>
#include <math.h>

// ---------- WMMA types (CDNA5 / gfx1250, wave32) ----------
typedef __bf16 v16bf __attribute__((ext_vector_type(16)));
typedef float  v8f   __attribute__((ext_vector_type(8)));

#define S_DIM 64
#define C_DIM 32
#define E_DIM 128
#define KW 7
#define SLICE (S_DIM * S_DIM * S_DIM)   // 262144
#define NVOX  (2 * SLICE)               // 524288

__device__ __forceinline__ unsigned short f2bfbits(float f) {
    unsigned u = __builtin_bit_cast(unsigned, f);
    unsigned r = 0x7FFFu + ((u >> 16) & 1u);       // round-to-nearest-even
    return (unsigned short)((u + r) >> 16);
}
__device__ __forceinline__ __bf16 bits2bf(unsigned short b) {
    return __builtin_bit_cast(__bf16, b);
}
// Branch-free tanh-form GELU: v_exp_f32 + v_rcp_f32, no EXEC divergence.
// tanh(u) = 1 - 2/(exp(2u)+1); large |u| saturates cleanly through inf/0.
__device__ __forceinline__ float gelu_fast(float z) {
    float u  = 0.7978845608028654f * fmaf(0.044715f * z * z, z, z);
    float e  = __expf(2.0f * u);
    float th = 1.0f - __fdividef(2.0f, e + 1.0f);
    return 0.5f * z * (1.0f + th);
}

// =====================================================================
// Kernel A: submanifold depthwise 7x7x7 conv (+bias, output-masked).
// One block = one (b,c) pair and an 8x8x8 output tile; 14^3 input halo
// tile (pre-masked => xm) and the 343 weights staged in LDS.
// Output written voxel-major to workspace: yws[vox*32 + c].
// =====================================================================
__global__ __launch_bounds__(256)
void dwconv_kernel(const float* __restrict__ x,
                   const int*   __restrict__ mask,
                   const float* __restrict__ w_dw,
                   const float* __restrict__ b_dw,
                   float* __restrict__ yws) {
    __shared__ float xt[14 * 14 * 14];   // masked input tile
    __shared__ float wl[KW * KW * KW];

    const int tile = blockIdx.x;                 // 0..511
    const int bc   = blockIdx.y;                 // 0..63
    const int b    = bc >> 5;
    const int c    = bc & 31;
    const int tx   = (tile & 7) * 8;
    const int ty   = ((tile >> 3) & 7) * 8;
    const int tz   = (tile >> 6) * 8;
    const int tid  = threadIdx.x;

    const float* xc = x + (size_t)bc * SLICE;
    const int*   mb = mask + (size_t)b * SLICE;

    // Stage masked input halo tile (zero-padded at volume edges).
    for (int i = tid; i < 14 * 14 * 14; i += 256) {
        int lz = i / 196, r = i % 196, ly = r / 14, lx = r % 14;
        int gz = tz + lz - 3, gy = ty + ly - 3, gx = tx + lx - 3;
        float v = 0.0f;
        if ((unsigned)gz < S_DIM && (unsigned)gy < S_DIM && (unsigned)gx < S_DIM) {
            int s = gz * 4096 + gy * 64 + gx;
            v = mb[s] ? xc[s] : 0.0f;            // xm = x * mask
        }
        xt[i] = v;
    }
    for (int i = tid; i < 343; i += 256) wl[i] = w_dw[c * 343 + i];
    __syncthreads();

    const float bias = b_dw[c];
    #pragma unroll
    for (int rep = 0; rep < 2; ++rep) {
        int o  = tid + rep * 256;                // 0..511
        int oz = o >> 6, oy = (o >> 3) & 7, ox = o & 7;
        int gz = tz + oz, gy = ty + oy, gx = tx + ox;
        int s  = gz * 4096 + gy * 64 + gx;
        int vox = b * SLICE + s;
        float acc = 0.0f;
        if (mb[s]) {                             // emit only at active sites
            acc = bias;
            for (int dz = 0; dz < KW; ++dz)
                for (int dy = 0; dy < KW; ++dy) {
                    const float* xrow = &xt[(oz + dz) * 196 + (oy + dy) * 14 + ox];
                    const float* wrow = &wl[(dz * KW + dy) * KW];
                    #pragma unroll
                    for (int dx = 0; dx < KW; ++dx)
                        acc = fmaf(xrow[dx], wrow[dx], acc);
                }
        }
        yws[(size_t)vox * C_DIM + c] = acc;
    }
}

// =====================================================================
// Kernel B: LN -> (bf16) WMMA expand -> GELU -> WMMA project ->
//           bias + residual + mask.  128 threads = 4 waves; each wave32
//           owns one 16-voxel tile (M=16).  Weights bf16-staged in LDS.
// y tile staged with gfx1250 async-to-LDS DMA (ASYNCcnt path).
// =====================================================================
__global__ __launch_bounds__(128)
void mlp_kernel(const float* __restrict__ x,
                const int*   __restrict__ mask,
                const float* __restrict__ yws,
                const float* __restrict__ ln_g,
                const float* __restrict__ ln_b,
                const float* __restrict__ w2,
                const float* __restrict__ b2,
                const float* __restrict__ w3,
                const float* __restrict__ b3,
                float* __restrict__ out) {
    __shared__ unsigned short w2s[E_DIM * C_DIM];      // w2[e][c] bf16
    __shared__ unsigned short w3s[C_DIM * E_DIM];      // w3[c][e] bf16
    __shared__ float ylds[4][16][C_DIM];
    __shared__ float olds[4][16][C_DIM];               // projected tile
    __shared__ float mu_s[4][16], rs_s[4][16], mf_s[4][16];
    __shared__ unsigned short hlds[4][16][E_DIM];      // gelu(h) bf16

    const int tid  = threadIdx.x;
    const int wave = tid >> 5;
    const int lane = tid & 31;
    const int lm   = lane & 15;        // M (A,row) or N (B,col) within tile
    const int kh   = lane >> 4;        // lane-half selects K sub-range

    __builtin_prefetch(w2, 0, 3);      // global_prefetch_b8 (gfx1250)
    for (int i = tid; i < E_DIM * C_DIM; i += 128) w2s[i] = f2bfbits(w2[i]);
    for (int i = tid; i < C_DIM * E_DIM; i += 128) w3s[i] = f2bfbits(w3[i]);

    const int tileVox = blockIdx.x * 64 + wave * 16;

    // ---- Async DMA: stage y tile [16 vox x 32 ch] = 2 KiB per wave.
    // GVS mode: SGPR base + per-lane 32-bit byte offset; LDS destination
    // offset carried in the VDST VGPR (generic LDS addr low 32 bits).
    {
        unsigned lbase = (unsigned)(size_t)(&ylds[wave][0][0]);
        unsigned long long gbase =
            (unsigned long long)(yws + (size_t)tileVox * C_DIM);
        #pragma unroll
        for (int it = 0; it < 4; ++it) {
            unsigned loff = lbase + (unsigned)(it * 512 + lane * 16);
            unsigned goff = (unsigned)(it * 512 + lane * 16);
            asm volatile("global_load_async_to_lds_b128 %0, %1, %2"
                         :: "v"(loff), "v"(goff), "s"(gbase)
                         : "memory");
        }
    }
    if (lane < 16) mf_s[wave][lane] = (float)mask[tileVox + lane];
    asm volatile("s_wait_asynccnt 0" ::: "memory");
    __syncthreads();

    // LayerNorm stats, one voxel per lane (lanes 0..15).
    if (lane < 16) {
        float mu = 0.0f;
        #pragma unroll
        for (int c = 0; c < C_DIM; ++c) mu += ylds[wave][lane][c];
        mu *= (1.0f / C_DIM);
        float var = 0.0f;
        #pragma unroll
        for (int c = 0; c < C_DIM; ++c) {
            float d = ylds[wave][lane][c] - mu;
            var = fmaf(d, d, var);
        }
        var *= (1.0f / C_DIM);
        mu_s[wave][lane] = mu;
        rs_s[wave][lane] = rsqrtf(var + 1e-6f);
    }
    __syncthreads();

    // --- A fragment: yn tile, 16x32 bf16, contiguous K layout:
    // lanes 0-15: elems 0-15 -> K0-15 ; lanes 16-31: K16-31.
    const float mu = mu_s[wave][lm];
    const float rs = rs_s[wave][lm];
    const float mf = mf_s[wave][lm];
    v16bf afrag;
    #pragma unroll
    for (int i = 0; i < 16; ++i) {
        int k = i + 16 * kh;
        float yn = fmaf((ylds[wave][lm][k] - mu) * rs, ln_g[k], ln_b[k]) * mf;
        afrag[i] = (__bf16)yn;
    }

    // --- Expand: 8 N-tiles of 16 over E=128, single K=32 step each.
    #pragma unroll
    for (int et = 0; et < 8; ++et) {
        v16bf bfrag;
        const int e = et * 16 + lm;
        #pragma unroll
        for (int i = 0; i < 16; ++i)                  // B: K = kh*16 + i
            bfrag[i] = bits2bf(w2s[e * C_DIM + kh * 16 + i]);
        v8f acc = {};
        acc = __builtin_amdgcn_wmma_f32_16x16x32_bf16(
            false, afrag, false, bfrag, (short)0, acc, false, false);
        #pragma unroll
        for (int r = 0; r < 8; ++r) {                 // C/D: M = r + 8*kh
            float z = acc[r] + b2[e];
            hlds[wave][r + 8 * kh][e] = f2bfbits(gelu_fast(z));
        }
    }
    __syncthreads();

    // --- Project: A = h (16x128), K in 4 steps of 32; N = 32 (2 tiles).
    v8f acc0 = {}, acc1 = {};
    #pragma unroll
    for (int ks = 0; ks < 4; ++ks) {
        v16bf ah;
        #pragma unroll
        for (int i = 0; i < 16; ++i) {
            int k = ks * 32 + i + 16 * kh;
            ah[i] = bits2bf(hlds[wave][lm][k]);
        }
        v16bf b0, b1;
        #pragma unroll
        for (int i = 0; i < 16; ++i) {
            int k = ks * 32 + kh * 16 + i;
            b0[i] = bits2bf(w3s[lm * E_DIM + k]);         // c = lm
            b1[i] = bits2bf(w3s[(16 + lm) * E_DIM + k]);  // c = 16 + lm
        }
        acc0 = __builtin_amdgcn_wmma_f32_16x16x32_bf16(
            false, ah, false, b0, (short)0, acc0, false, false);
        acc1 = __builtin_amdgcn_wmma_f32_16x16x32_bf16(
            false, ah, false, b1, (short)0, acc1, false, false);
    }

    // Stash projected tile (+b3) to LDS for coalesced residual writes.
    #pragma unroll
    for (int nt = 0; nt < 2; ++nt) {
        const int c = nt * 16 + lm;
        #pragma unroll
        for (int r = 0; r < 8; ++r)
            olds[wave][r + 8 * kh][c] = (nt == 0 ? acc0[r] : acc1[r]) + b3[c];
    }
    __syncthreads();

    // --- Epilogue: out = m * (x + o), lanes walk x-axis => 64B segments.
    {
        const int vz  = lm;                 // voxel within tile
        const int ch2 = kh;                 // channel parity
        const int vox = tileVox + vz;
        const int b   = vox >> 18;          // / 262144
        const int s   = vox & (SLICE - 1);
        const float mfv = mf_s[wave][vz];
        #pragma unroll
        for (int j = 0; j < 16; ++j) {
            const int c = j * 2 + ch2;
            size_t oi = ((size_t)(b * C_DIM + c)) * SLICE + s;
            out[oi] = mfv * (x[oi] + olds[wave][vz][c]);
        }
    }
}

extern "C" void kernel_launch(void* const* d_in, const int* in_sizes, int n_in,
                              void* d_out, int out_size, void* d_ws, size_t ws_size,
                              hipStream_t stream) {
    const float* x    = (const float*)d_in[0];
    const int*   mask = (const int*)  d_in[1];
    const float* w_dw = (const float*)d_in[2];
    const float* b_dw = (const float*)d_in[3];
    const float* ln_g = (const float*)d_in[4];
    const float* ln_b = (const float*)d_in[5];
    const float* w2   = (const float*)d_in[6];
    const float* b2   = (const float*)d_in[7];
    const float* w3   = (const float*)d_in[8];
    const float* b3   = (const float*)d_in[9];
    float* out = (float*)d_out;
    float* yws = (float*)d_ws;                     // [NVOX x 32] f32 = 64 MiB

    // Kernel A: 512 spatial tiles x 64 (b,c) pairs, 256 threads.
    dwconv_kernel<<<dim3(512, 64), 256, 0, stream>>>(x, mask, w_dw, b_dw, yws);
    // Kernel B: 8192 blocks x 128 threads (4 waves x 16-voxel tiles).
    mlp_kernel<<<dim3(NVOX / 64), 128, 0, stream>>>(
        x, mask, yws, ln_g, ln_b, w2, b2, w3, b3, out);
}
